// SequenceToPairBlock_82678120448112
// MI455X (gfx1250) — matrix operand order repaired and checked by reading, not compiled, not run on hardware
//
#include <hip/hip_runtime.h>
#include <hip/hip_bf16.h>

typedef __attribute__((ext_vector_type(16))) __bf16 v16bf;
typedef __attribute__((ext_vector_type(8)))  float  v8f;

#define SEQ 512            // pooled sequence length
#define HEADS 32
#define CDIM 128
#define HC 4096
#define DDIM 1536
#define PAIRF 128

// ---------------------------------------------------------------------------
// WMMA helpers (bf16 A/B, f32 accumulate), layouts per CDNA5 ISA 7.12.2
// ---------------------------------------------------------------------------
__device__ inline v8f wmma_bf16(v16bf a, v16bf b, v8f c) {
  return __builtin_amdgcn_wmma_f32_16x16x32_bf16(false, a, false, b, (short)0, c,
                                                 false, false);
}

// A-matrix 16x32 bf16 fragment from row-major [ld] matrix.
// lane<16: row=lane, K = {kh..kh+7, kh+16..kh+23} with kh=0 ; lane>=16: kh=8.
__device__ inline v16bf load_frag_a(const __bf16* mat, size_t ld, int row0,
                                    int coloff, int lane) {
  int m  = lane & 15;
  int kh = (lane >> 4) << 3;
  const __bf16* p = mat + (size_t)(row0 + m) * ld + coloff + kh;
  union { v16bf v; uint4 u[2]; } t;
  t.u[0] = *(const uint4*)(p);
  t.u[1] = *(const uint4*)(p + 16);
  return t.v;
}

// B-matrix 32x16 bf16 fragment; source stored "transposed": each output
// column n is a row of `mat` (K contiguous).  K = {kh..kh+7, kh+16..kh+23}, kh=0 (lane<16) / 8 (lane>=16).
__device__ inline v16bf load_frag_b(const __bf16* mat, size_t ld, int coloff,
                                    int rowbase, int rowmax, int lane) {
  int n  = lane & 15;
  int kb = (lane >> 4) << 3;
  int r  = rowbase + n;
  if (r < 0) r = 0;
  if (r > rowmax) r = rowmax;
  const __bf16* p = mat + (size_t)r * ld + coloff + kb;
  union { v16bf v; uint4 u[2]; } t;
  t.u[0] = *(const uint4*)(p);
  t.u[1] = *(const uint4*)(p + 16);
  return t.v;
}

// ---------------------------------------------------------------------------
// 1) transpose fp32 (R,C) -> bf16 (C,R)
// ---------------------------------------------------------------------------
__global__ void transpose_bf16_kernel(const float* __restrict__ src,
                                      __bf16* __restrict__ dst, int R, int C) {
  size_t idx = (size_t)blockIdx.x * blockDim.x + threadIdx.x;
  if (idx >= (size_t)R * C) return;
  int r = (int)(idx / C), c = (int)(idx % C);
  dst[(size_t)c * R + r] = (__bf16)src[idx];
}

// ---------------------------------------------------------------------------
// 2) pool(16) + RMSNorm*scale + GELU(tanh)  ->  xp_bf16, g_bf16
// ---------------------------------------------------------------------------
__global__ void pool_norm_kernel(const float* __restrict__ x,
                                 const float* __restrict__ scale,
                                 __bf16* __restrict__ xp_out,
                                 __bf16* __restrict__ g_out) {
  const int row = blockIdx.x;             // b*SEQ + s, 1024 rows
  const int tid = threadIdx.x;            // 256 threads
  const float* xb = x + (size_t)row * 16 * DDIM;
  float pooled[6];
  float ss = 0.f;
#pragma unroll
  for (int i = 0; i < 6; ++i) {
    int d = tid + i * 256;
    float acc = 0.f;
#pragma unroll
    for (int p = 0; p < 16; ++p) acc += xb[(size_t)p * DDIM + d];
    acc *= (1.0f / 16.0f);
    pooled[i] = acc;
    ss += acc * acc;
  }
  __shared__ float red[256];
  red[tid] = ss;
  __syncthreads();
  for (int s = 128; s > 0; s >>= 1) {
    if (tid < s) red[tid] += red[tid + s];
    __syncthreads();
  }
  float inv = rsqrtf(red[0] * (1.0f / (float)DDIM) + 1e-5f);
#pragma unroll
  for (int i = 0; i < 6; ++i) {
    int d = tid + i * 256;
    float v = pooled[i] * inv * scale[d];
    xp_out[(size_t)row * DDIM + d] = (__bf16)v;
    float g = 0.5f * v *
              (1.0f + tanhf(0.7978845608028654f * (v + 0.044715f * v * v * v)));
    g_out[(size_t)row * DDIM + d] = (__bf16)g;
  }
}

// ---------------------------------------------------------------------------
// 3) analytic position features (2S, 64) -> bf16
// ---------------------------------------------------------------------------
__global__ void pos_feat_kernel(__bf16* __restrict__ pf) {
  int idx = blockIdx.x * 256 + threadIdx.x;   // 1024 * 64
  if (idx >= 2 * SEQ * 64) return;
  int p = idx >> 6, f2 = idx & 63, f = f2 & 31;
  float rel = (float)(p - SEQ);
  float dist = fabsf(rel);
  // center_widths[f] = f + geomspace(1, 481, 32, endpoint=False)[f] = f + 481^(f/32)
  float cw = (float)f + powf(481.0f, (float)f * (1.0f / 32.0f));
  float v = (cw > dist) ? 1.0f : 0.0f;
  if (f2 >= 32) v *= (rel > 0.f) ? 1.f : ((rel < 0.f) ? -1.f : 0.f);
  pf[idx] = (__bf16)v;
}

// ---------------------------------------------------------------------------
// 4) WMMA GEMM: C[M,N] = A[M,K] (bf16) @ BT[N,K]^T (bf16)
//    Block = 4 waves; the 16x32 A tile is staged once per block into LDS via
//    global_load_async_to_lds_b128 (ASYNCcnt path), then each wave computes a
//    16x64 slice (4 accumulators reuse one A fragment).
//    epilogue: +bias -> {fp32 out | bf16 out | bf16 out with extra bias2}
// ---------------------------------------------------------------------------
__global__ void gemm_bf16_kernel(const __bf16* __restrict__ A,
                                 const __bf16* __restrict__ BT,
                                 int M, int N, int K,
                                 const float* __restrict__ bias,
                                 __bf16* __restrict__ out_bf16,
                                 const float* __restrict__ bias2,
                                 __bf16* __restrict__ out_bf16_biased,
                                 float* __restrict__ out_f32) {
  __shared__ __bf16 sA[16 * 32];        // 1 KB A tile shared by the block
  const int tid = threadIdx.x;
  const int wave = tid >> 5, lane = tid & 31;
  const int m0 = blockIdx.y * 16;
  const int nw = blockIdx.x * 256 + wave * 64;   // this wave's 64-col slice
  const v8f vzero = {0.f, 0.f, 0.f, 0.f, 0.f, 0.f, 0.f, 0.f};
  v8f acc[4] = {vzero, vzero, vzero, vzero};

  // async-copy assignment: threads 0..63 move one 16-byte chunk each
  const int cRow = tid >> 2;            // 0..15
  const int cCol = (tid & 3) << 3;      // 0,8,16,24 (bf16 elements)
  const uint32_t ldsDst =
      (uint32_t)(uintptr_t)(sA + cRow * 32 + cCol);  // addr[31:0] == LDS offset

  for (int k0 = 0; k0 < K; k0 += 32) {
    if (tid < 64) {
      uint64_t ga = (uint64_t)(uintptr_t)(A + (size_t)(m0 + cRow) * K + k0 + cCol);
      asm volatile("global_load_async_to_lds_b128 %0, %1, off"
                   :
                   : "v"(ldsDst), "v"(ga)
                   : "memory");
    }
    asm volatile("s_wait_asynccnt 0x0" ::: "memory");
    __syncthreads();

    // A fragment from LDS (shared across all 4 waves)
    v16bf a;
    {
      const __bf16* p = sA + (lane & 15) * 32 + ((lane >> 4) << 3);
      union { v16bf v; uint4 u[2]; } t;
      t.u[0] = *(const uint4*)(p);
      t.u[1] = *(const uint4*)(p + 16);
      a = t.v;
    }

    // prefetch next K-slice of this wave's B rows (global_prefetch_b8)
    if (k0 + 32 < K) {
      int pr = nw + (lane & 15);
      if (pr > N - 1) pr = N - 1;
      __builtin_prefetch(BT + (size_t)pr * K + k0 + 32, 0, 1);
    }

#pragma unroll
    for (int j = 0; j < 4; ++j) {
      v16bf b = load_frag_b(BT, (size_t)K, k0, nw + j * 16, N - 1, lane);
      acc[j] = wmma_bf16(a, b, acc[j]);
    }
    __syncthreads();   // before next iteration overwrites sA
  }

  const int mh = (lane >> 4) << 3;
#pragma unroll
  for (int j = 0; j < 4; ++j) {
    int n = nw + j * 16 + (lane & 15);
    if (n >= N) continue;
    float bval = bias ? bias[n] : 0.f;
#pragma unroll
    for (int v = 0; v < 8; ++v) {
      int m = m0 + mh + v;
      float val = acc[j][v] + bval;
      size_t o = (size_t)m * N + n;
      if (out_f32) out_f32[o] = val;
      if (out_bf16) out_bf16[o] = (__bf16)val;
      if (out_bf16_biased) out_bf16_biased[o] = (__bf16)(val + bias2[n]);
    }
  }
}

// ---------------------------------------------------------------------------
// 5) attention-logit tile kernel:
//    a[b,q,k,h] = q·k + 0.5*( qpb·pos[S+k-q] + kpb·pos[S+q-k] )   -> bf16
//    Shift handled via 16x32 WMMA against a 31-row pos window; the diagonal
//    element j = 15 + k - q is gathered through LDS.
// ---------------------------------------------------------------------------
__global__ void a_tile_kernel(const __bf16* __restrict__ qm,
                              const __bf16* __restrict__ km,
                              const __bf16* __restrict__ qpb,
                              const __bf16* __restrict__ kpb,
                              const __bf16* __restrict__ pos,
                              __bf16* __restrict__ a_out) {
  __shared__ float T1[4][16 * 32];
  __shared__ float T2[4][16 * 32];
  const int wave = threadIdx.x >> 5, lane = threadIdx.x & 31;
  const int k0 = blockIdx.x * 16;
  const int q0 = blockIdx.y * 16;
  const int b  = blockIdx.z;
  float* t1 = T1[wave];
  float* t2 = T2[wave];
  const int w0q = SEQ + k0 - q0 - 15;   // rel_q pos-window base (>=1)
  const int w0k = SEQ + q0 - k0 - 15;   // rel_k pos-window base (>=1)
  const int qrow0 = b * SEQ + q0;
  const int krow0 = b * SEQ + k0;
  const int ncol = lane & 15, mh = (lane >> 4) << 3;
  const v8f vzero = {0.f, 0.f, 0.f, 0.f, 0.f, 0.f, 0.f, 0.f};

  for (int hh = 0; hh < 8; ++hh) {
    const int h = wave * 8 + hh;
    const int cb = h * CDIM;

    // q . k^T  (16x16, K = 128)
    v8f acc = vzero;
#pragma unroll
    for (int c0 = 0; c0 < CDIM; c0 += 32) {
      v16bf a = load_frag_a(qm, HC, qrow0, cb + c0, lane);
      v16bf bb = load_frag_b(km, HC, cb + c0, krow0, 1 << 30, lane);
      acc = wmma_bf16(a, bb, acc);
    }

    // rel_q: (q+qb)_tile (16xC) @ pos_window^T (Cx32)
    v8f r0 = vzero, r1 = vzero;
#pragma unroll
    for (int c0 = 0; c0 < CDIM; c0 += 32) {
      v16bf a  = load_frag_a(qpb, HC, qrow0, cb + c0, lane);
      v16bf b0 = load_frag_b(pos, HC, cb + c0, w0q,      2 * SEQ - 1, lane);
      v16bf b1 = load_frag_b(pos, HC, cb + c0, w0q + 16, 2 * SEQ - 1, lane);
      r0 = wmma_bf16(a, b0, r0);
      r1 = wmma_bf16(a, b1, r1);
    }
#pragma unroll
    for (int v = 0; v < 8; ++v) {
      t1[(mh + v) * 32 + ncol]      = r0[v];
      t1[(mh + v) * 32 + 16 + ncol] = r1[v];
    }

    // rel_k: (k+kb)_tile (16xC) @ pos_window^T (Cx32)
    v8f s0 = vzero, s1 = vzero;
#pragma unroll
    for (int c0 = 0; c0 < CDIM; c0 += 32) {
      v16bf a  = load_frag_a(kpb, HC, krow0, cb + c0, lane);
      v16bf b0 = load_frag_b(pos, HC, cb + c0, w0k,      2 * SEQ - 1, lane);
      v16bf b1 = load_frag_b(pos, HC, cb + c0, w0k + 16, 2 * SEQ - 1, lane);
      s0 = wmma_bf16(a, b0, s0);
      s1 = wmma_bf16(a, b1, s1);
    }
#pragma unroll
    for (int v = 0; v < 8; ++v) {
      t2[(mh + v) * 32 + ncol]      = s0[v];
      t2[(mh + v) * 32 + 16 + ncol] = s1[v];
    }

    __builtin_amdgcn_wave_barrier();  // keep LDS RAW ordering within the wave

    // combine along the shift diagonal and store a[b,q,k,h]
#pragma unroll
    for (int v = 0; v < 8; ++v) {
      int m = mh + v, n = ncol;
      float val = acc[v] + 0.5f * (t1[m * 32 + (15 + n - m)] +
                                   t2[n * 32 + (15 + m - n)]);
      a_out[(((size_t)b * SEQ + q0 + m) * SEQ + (k0 + n)) * HEADS + h] =
          (__bf16)val;
    }
  }
}

// ---------------------------------------------------------------------------
// 6) pair_act = a @ w_pair (K = 32, one WMMA) + b_pair + y_q + y_k  (fp32 out)
// ---------------------------------------------------------------------------
__global__ void pair_out_kernel(const __bf16* __restrict__ a_mat,
                                const __bf16* __restrict__ wpT,
                                const float* __restrict__ b_pair,
                                const float* __restrict__ y_q,
                                const float* __restrict__ y_k,
                                float* __restrict__ out) {
  const int wave = threadIdx.x >> 5, lane = threadIdx.x & 31;
  const int n0 = blockIdx.y * 64 + wave * 16;
  const int m0 = blockIdx.x * 16;   // row tile over B*SEQ*SEQ
  v16bf a = load_frag_a(a_mat, HEADS, m0, 0, lane);
  v16bf b = load_frag_b(wpT, HEADS, 0, n0, PAIRF - 1, lane);
  const v8f vzero = {0.f, 0.f, 0.f, 0.f, 0.f, 0.f, 0.f, 0.f};
  v8f acc = wmma_bf16(a, b, vzero);
  const int rrem = m0 % (SEQ * SEQ);
  const int bb = m0 / (SEQ * SEQ);
  const int qq = rrem / SEQ;
  const int kk0 = rrem % SEQ;
  const int f = n0 + (lane & 15);
  const int mh = (lane >> 4) << 3;
  float base = b_pair[f] + y_q[(size_t)(bb * SEQ + qq) * PAIRF + f];
#pragma unroll
  for (int v = 0; v < 8; ++v) {
    int m = mh + v;
    float val = acc[v] + base + y_k[(size_t)(bb * SEQ + kk0 + m) * PAIRF + f];
    out[(size_t)(m0 + m) * PAIRF + f] = val;
  }
}

// ---------------------------------------------------------------------------
extern "C" void kernel_launch(void* const* d_in, const int* in_sizes, int n_in,
                              void* d_out, int out_size, void* d_ws,
                              size_t ws_size, hipStream_t stream) {
  const float* x        = (const float*)d_in[0];
  const float* scale    = (const float*)d_in[1];
  const float* w_q      = (const float*)d_in[2];
  const float* w_k      = (const float*)d_in[3];
  const float* w_pos    = (const float*)d_in[4];
  const float* b_pos    = (const float*)d_in[5];
  const float* q_r_bias = (const float*)d_in[6];   // (H*C) broadcast bias
  const float* k_r_bias = (const float*)d_in[7];
  const float* w_yq     = (const float*)d_in[8];
  const float* w_yk     = (const float*)d_in[9];
  const float* w_pair   = (const float*)d_in[10];
  const float* b_pair   = (const float*)d_in[11];
  float* out = (float*)d_out;

  char* ws = (char*)d_ws;
  size_t off = 0;
  auto alloc = [&](size_t bytes) -> void* {
    void* p = ws + off;
    off += (bytes + 255) & ~(size_t)255;
    return p;
  };
  const int BS = 2 * SEQ;  // 1024 rows
  __bf16* xp_bf   = (__bf16*)alloc((size_t)BS * DDIM * 2);
  __bf16* g_bf    = (__bf16*)alloc((size_t)BS * DDIM * 2);
  __bf16* wqT     = (__bf16*)alloc((size_t)HC * DDIM * 2);
  __bf16* wkT     = (__bf16*)alloc((size_t)HC * DDIM * 2);
  __bf16* wposT   = (__bf16*)alloc((size_t)HC * 64 * 2);
  __bf16* wyqT    = (__bf16*)alloc((size_t)PAIRF * DDIM * 2);
  __bf16* wykT    = (__bf16*)alloc((size_t)PAIRF * DDIM * 2);
  __bf16* wpairT  = (__bf16*)alloc((size_t)PAIRF * HEADS * 2);
  __bf16* posfeat = (__bf16*)alloc((size_t)(2 * SEQ) * 64 * 2);
  __bf16* q_bf    = (__bf16*)alloc((size_t)BS * HC * 2);
  __bf16* qpb_bf  = (__bf16*)alloc((size_t)BS * HC * 2);
  __bf16* k_bf    = (__bf16*)alloc((size_t)BS * HC * 2);
  __bf16* kpb_bf  = (__bf16*)alloc((size_t)BS * HC * 2);
  __bf16* pose_bf = (__bf16*)alloc((size_t)(2 * SEQ) * HC * 2);
  float*  y_q     = (float*)alloc((size_t)BS * PAIRF * 4);
  float*  y_k     = (float*)alloc((size_t)BS * PAIRF * 4);
  __bf16* a_bf    = (__bf16*)alloc((size_t)2 * SEQ * SEQ * HEADS * 2);

  auto gridFor = [](size_t n) { return (unsigned)((n + 255) / 256); };

  // weight transposes (fp32 -> bf16, (R,C)->(C,R))
  transpose_bf16_kernel<<<gridFor((size_t)DDIM * HC), 256, 0, stream>>>(w_q, wqT, DDIM, HC);
  transpose_bf16_kernel<<<gridFor((size_t)DDIM * HC), 256, 0, stream>>>(w_k, wkT, DDIM, HC);
  transpose_bf16_kernel<<<gridFor((size_t)64 * HC), 256, 0, stream>>>(w_pos, wposT, 64, HC);
  transpose_bf16_kernel<<<gridFor((size_t)DDIM * PAIRF), 256, 0, stream>>>(w_yq, wyqT, DDIM, PAIRF);
  transpose_bf16_kernel<<<gridFor((size_t)DDIM * PAIRF), 256, 0, stream>>>(w_yk, wykT, DDIM, PAIRF);
  transpose_bf16_kernel<<<gridFor((size_t)HEADS * PAIRF), 256, 0, stream>>>(w_pair, wpairT, HEADS, PAIRF);

  // pooling + rmsnorm + gelu ; position features
  pool_norm_kernel<<<BS, 256, 0, stream>>>(x, scale, xp_bf, g_bf);
  pos_feat_kernel<<<gridFor((size_t)2 * SEQ * 64), 256, 0, stream>>>(posfeat);

  // q / k projections (also store rel-bias-added bf16 copies)
  gemm_bf16_kernel<<<dim3(HC / 256, BS / 16), 128, 0, stream>>>(
      xp_bf, wqT, BS, HC, DDIM, nullptr, q_bf, q_r_bias, qpb_bf, nullptr);
  gemm_bf16_kernel<<<dim3(HC / 256, BS / 16), 128, 0, stream>>>(
      xp_bf, wkT, BS, HC, DDIM, nullptr, k_bf, k_r_bias, kpb_bf, nullptr);
  // position encoding (2S, HC) = posfeat @ w_pos + b_pos
  gemm_bf16_kernel<<<dim3(HC / 256, (2 * SEQ) / 16), 128, 0, stream>>>(
      posfeat, wposT, 2 * SEQ, HC, 64, b_pos, pose_bf, nullptr, nullptr, nullptr);
  // y_q / y_k = gelu(xp) @ w_y{q,k}  (fp32)
  gemm_bf16_kernel<<<dim3(1, BS / 16), 128, 0, stream>>>(
      g_bf, wyqT, BS, PAIRF, DDIM, nullptr, nullptr, nullptr, nullptr, y_q);
  gemm_bf16_kernel<<<dim3(1, BS / 16), 128, 0, stream>>>(
      g_bf, wykT, BS, PAIRF, DDIM, nullptr, nullptr, nullptr, nullptr, y_k);

  // attention logits with shifted relative-position terms
  a_tile_kernel<<<dim3(SEQ / 16, SEQ / 16, 2), 128, 0, stream>>>(
      q_bf, k_bf, qpb_bf, kpb_bf, pose_bf, a_bf);

  // final pair activation
  pair_out_kernel<<<dim3((2u * SEQ * SEQ) / 16, PAIRF / 64), 128, 0, stream>>>(
      a_bf, wpairT, b_pair, y_q, y_k, out);
}
